// TabRM_53721450939148
// MI455X (gfx1250) — matrix operand order repaired and checked by reading, not compiled, not run on hardware
//
#include <hip/hip_runtime.h>
#include <hip/hip_bf16.h>

// ---------------------------------------------------------------------------
// MI455X (gfx1250) pipeline for the TabR-style retrieval reference.
// Precision: bf16 operands + f32 accumulate via v_wmma_f32_16x16x32_bf16.
// Staging:  GLOBAL_LOAD_ASYNC_TO_LDS_B128 (ASYNCcnt) + double-buffered LDS.
// ---------------------------------------------------------------------------

typedef __attribute__((ext_vector_type(16))) __bf16 bf16x16;
typedef __attribute__((ext_vector_type(8)))  float  v8f;

#define EPSF 1e-5f

__device__ __forceinline__ unsigned short f2bfu(float f) {
    unsigned int u = __builtin_bit_cast(unsigned int, f);
    unsigned int r = u + 0x7FFFu + ((u >> 16) & 1u);   // round-to-nearest-even
    return (unsigned short)(r >> 16);
}
__device__ __forceinline__ __bf16 f2bf(float f) {
    unsigned short s = f2bfu(f);
    return __builtin_bit_cast(__bf16, s);
}
__device__ __forceinline__ float bf2f(__bf16 h) {
    unsigned int u = ((unsigned int)__builtin_bit_cast(unsigned short, h)) << 16;
    return __builtin_bit_cast(float, u);
}

// Async cache->LDS copy of 16 bytes, tracked by ASYNCcnt (gfx1250).
// lds_off = low 32 bits of the flat address of an LDS location (aperture
// layout: addr[31:0] IS the LDS byte offset). gaddr = 64-bit global address.
__device__ __forceinline__ void async_copy_b128(unsigned int lds_off, unsigned long long gaddr) {
#if defined(__gfx1250__)
    asm volatile("global_load_async_to_lds_b128 %0, %1, off"
                 :: "v"(lds_off), "v"(gaddr) : "memory");
#else
    (void)lds_off; (void)gaddr;
#endif
}
__device__ __forceinline__ void wait_async0() {
#if defined(__gfx1250__)
    asm volatile("s_wait_asynccnt 0x0" ::: "memory");
#endif
}

// ---------------------------------------------------------------------------
// fp32 -> bf16 convert (grid-stride)
// ---------------------------------------------------------------------------
__global__ void k_f2bf(const float* __restrict__ in, __bf16* __restrict__ out, size_t n) {
    size_t i  = (size_t)blockIdx.x * blockDim.x + threadIdx.x;
    size_t st = (size_t)gridDim.x * blockDim.x;
    for (; i < n; i += st) out[i] = f2bf(in[i]);
}

// ---------------------------------------------------------------------------
// WMMA GEMM: C[M,N] = epilogue( A[M,K] * W[N,K]^T )
//   block tile 128x128, K-step 32, 256 threads = 8 waves, wave tile 64x32.
//   Double-buffered LDS fed by async B128 copies; WMMA overlaps the async
//   transfer of the next K-tile.
// Epilogues: 0=bias, 1=bias+relu, 2=bias+BN, 3=distance (rowv[m]-2*acc+bias[n])
// ---------------------------------------------------------------------------
static constexpr int EP_BIAS = 0, EP_BIAS_RELU = 1, EP_BN = 2, EP_DIST = 3;

template<int EPIL, bool OUT_BF16>
__global__ void __launch_bounds__(256) k_gemm(
    const __bf16* __restrict__ A, const __bf16* __restrict__ W,
    int M, int Nout, int Kdim,
    const float* __restrict__ bias,  // per-n bias (or candidate norms for DIST)
    const float* __restrict__ gg, const float* __restrict__ bb,
    const float* __restrict__ mm, const float* __restrict__ vv,   // BN params
    const float* __restrict__ rowv,                                // per-m (query norms)
    void* __restrict__ Cout)
{
    // 40 = 32 + 8 pad: conflict-free fragment reads, rows stay 16B aligned (80B)
    __shared__ __align__(16) __bf16 sA[2][128 * 40];
    __shared__ __align__(16) __bf16 sB[2][128 * 40];

    const int m0   = blockIdx.y * 128;
    const int n0   = blockIdx.x * 128;
    const int tid  = threadIdx.x;
    const int lrow = tid >> 1;          // 0..127 : tile row loaded by this thread
    const int lcol = (tid & 1) * 16;    // bf16 offset 0 / 16 within the 32-wide K row
    const int wave = tid >> 5;
    const int lane = tid & 31;
    const int wm   = wave >> 2;         // 0..1  -> 64-row slab
    const int wn   = wave & 3;          // 0..3  -> 32-col slab
    const int ln   = lane & 15;
    const int lhi  = lane >> 4;

    v8f acc[4][2];
    const v8f vzero = {0.f,0.f,0.f,0.f,0.f,0.f,0.f,0.f};
    #pragma unroll
    for (int i = 0; i < 4; ++i)
        #pragma unroll
        for (int j = 0; j < 2; ++j) acc[i][j] = vzero;

    const __bf16* gA = A + (size_t)(m0 + lrow) * Kdim + lcol;
    const __bf16* gB = W + (size_t)(n0 + lrow) * Kdim + lcol;

    const unsigned int laBase[2] = {
        (unsigned int)(size_t)(&sA[0][lrow * 40 + lcol]),
        (unsigned int)(size_t)(&sA[1][lrow * 40 + lcol]) };
    const unsigned int lbBase[2] = {
        (unsigned int)(size_t)(&sB[0][lrow * 40 + lcol]),
        (unsigned int)(size_t)(&sB[1][lrow * 40 + lcol]) };

    auto issue_tile = [&](int kt, int b) {
        unsigned long long ga = (unsigned long long)(size_t)(gA + kt);
        unsigned long long gb = (unsigned long long)(size_t)(gB + kt);
        async_copy_b128(laBase[b],      ga);
        async_copy_b128(laBase[b] + 16, ga + 16);
        async_copy_b128(lbBase[b],      gb);
        async_copy_b128(lbBase[b] + 16, gb + 16);
    };

    issue_tile(0, 0);
    int buf = 0;
    for (int kt = 0; kt < Kdim; kt += 32, buf ^= 1) {
        wait_async0();        // this wave's async copies for `buf` landed in LDS
        __syncthreads();      // all waves' copies landed; spare buffer free to reuse
        if (kt + 32 < Kdim) {
            issue_tile(kt + 32, buf ^ 1);           // overlap with WMMA below
            if (kt + 64 < Kdim) {                   // keep L2 warm two tiles ahead
                __builtin_prefetch(gA + kt + 64, 0, 1);
                __builtin_prefetch(gB + kt + 64, 0, 1);
            }
        }

        // B fragments: lane = N (ln); regs 0..3 -> K = lhi*8 + {0..7}, regs 4..7 -> K = lhi*8 + 16 + {0..7}
        union { unsigned int u[8]; bf16x16 v; } bf_[2], af[4];
        #pragma unroll
        for (int nt = 0; nt < 2; ++nt) {
            const __bf16* p = &sB[buf][(wn * 32 + nt * 16 + ln) * 40 + lhi * 8];
            #pragma unroll
            for (int v = 0; v < 8; ++v) bf_[nt].u[v] = *(const unsigned int*)(p + 2 * v + (v >= 4 ? 8 : 0));
        }
        // A fragments: lane = M (ln); regs 0..3 -> K = lhi*8 + {0..7},
        //                             regs 4..7 -> K = lhi*8 + 16 + {0..7}
        #pragma unroll
        for (int mt = 0; mt < 4; ++mt) {
            const __bf16* p = &sA[buf][(wm * 64 + mt * 16 + ln) * 40 + lhi * 8];
            #pragma unroll
            for (int v = 0; v < 4; ++v) af[mt].u[v]     = *(const unsigned int*)(p + 2 * v);
            #pragma unroll
            for (int v = 0; v < 4; ++v) af[mt].u[4 + v] = *(const unsigned int*)(p + 16 + 2 * v);
        }
        #pragma unroll
        for (int mt = 0; mt < 4; ++mt)
            #pragma unroll
            for (int nt = 0; nt < 2; ++nt)
                acc[mt][nt] = __builtin_amdgcn_wmma_f32_16x16x32_bf16(
                    false, af[mt].v, false, bf_[nt].v,
                    (short)0, acc[mt][nt], false, false);

        __syncthreads();      // all waves done reading `buf` before it is refilled
    }

    // Epilogue. C/D layout: lane = N (ln), VGPR r -> M = r + 8*lhi.
    #pragma unroll
    for (int nt = 0; nt < 2; ++nt) {
        const int n = n0 + wn * 32 + nt * 16 + ln;
        float bn = bias ? bias[n] : 0.f;
        float sc = 1.f, sh = bn;
        if (EPIL == EP_BN) {
            float rs = rsqrtf(vv[n] + EPSF);
            sc = gg[n] * rs;                       // out = sc*acc + sh
            sh = sc * (bn - mm[n]) + bb[n];
        }
        #pragma unroll
        for (int mt = 0; mt < 4; ++mt) {
            const int mbase = m0 + wm * 64 + mt * 16 + 8 * lhi;
            #pragma unroll
            for (int r = 0; r < 8; ++r) {
                const int m = mbase + r;
                float val = acc[mt][nt][r];
                if (EPIL == EP_BIAS)           val += bn;
                else if (EPIL == EP_BIAS_RELU) { val += bn; val = val > 0.f ? val : 0.f; }
                else if (EPIL == EP_BN)        val = sc * val + sh;
                else /* EP_DIST */             val = rowv[m] - 2.f * val + bn;
                if (OUT_BF16) ((__bf16*)Cout)[(size_t)m * Nout + n] = f2bf(val);
                else          ((float*) Cout)[(size_t)m * Nout + n] = val;
            }
        }
    }
}

// ---------------------------------------------------------------------------
// Row squared-norm of a bf16 matrix (one wave32 per row)
// ---------------------------------------------------------------------------
__global__ void __launch_bounds__(256) k_rownorm(const __bf16* __restrict__ e,
                                                 float* __restrict__ out,
                                                 int rows, int cols) {
    int wave = threadIdx.x >> 5, lane = threadIdx.x & 31;
    int row = blockIdx.x * 8 + wave;
    if (row >= rows) return;
    const __bf16* p = e + (size_t)row * cols;
    float s = 0.f;
    for (int i = lane; i < cols; i += 32) { float f = bf2f(p[i]); s += f * f; }
    #pragma unroll
    for (int o = 16; o > 0; o >>= 1) s += __shfl_down(s, o, 32);
    if (lane == 0) out[row] = s;
}

// ---------------------------------------------------------------------------
// Top-32 pass 1: per (query, chunk of 4096) argmin-extract 32 into partials
// ---------------------------------------------------------------------------
__global__ void __launch_bounds__(128) k_topk_part(const float* __restrict__ d2, int Ncand,
                                                   float* __restrict__ pv, int* __restrict__ pi) {
    __shared__ float sv[4096];
    __shared__ float rv[128];
    __shared__ int   ri[128];
    const int q = blockIdx.y, c = blockIdx.x, t = threadIdx.x;
    const size_t base = (size_t)q * Ncand + (size_t)c * 4096;
    for (int j = t; j < 4096; j += 128) sv[j] = d2[base + j];
    __syncthreads();
    const int obase = (q * (int)gridDim.x + c) * 32;
    for (int k = 0; k < 32; ++k) {
        float bv = 3.0e38f; int bi = 0x7fffffff;
        for (int j = t; j < 4096; j += 128) {
            float v = sv[j];
            if (v < bv || (v == bv && j < bi)) { bv = v; bi = j; }
        }
        rv[t] = bv; ri[t] = bi;
        __syncthreads();
        for (int s = 64; s > 0; s >>= 1) {
            if (t < s) {
                float ov = rv[t + s]; int oi = ri[t + s];
                if (ov < rv[t] || (ov == rv[t] && oi < ri[t])) { rv[t] = ov; ri[t] = oi; }
            }
            __syncthreads();
        }
        if (t == 0) {
            int w = ri[0];
            pv[obase + k] = rv[0];
            pi[obase + k] = c * 4096 + w;
            sv[w] = 3.0e38f;
        }
        __syncthreads();
    }
}

// ---------------------------------------------------------------------------
// Top-32 pass 2: merge 32 chunks x 32 partials -> final 32 indices per query
// ---------------------------------------------------------------------------
__global__ void __launch_bounds__(128) k_topk_merge(const float* __restrict__ pv,
                                                    const int* __restrict__ pi,
                                                    int nper, int* __restrict__ outIdx) {
    __shared__ float sv[1024];
    __shared__ int   si[1024];
    __shared__ float rv[128];
    __shared__ int   ri[128];
    const int q = blockIdx.x, t = threadIdx.x;
    for (int j = t; j < nper; j += 128) {
        sv[j] = pv[(size_t)q * nper + j];
        si[j] = pi[(size_t)q * nper + j];
    }
    __syncthreads();
    for (int k = 0; k < 32; ++k) {
        float bv = 3.0e38f; int bj = 0; int bc = 0x7fffffff;
        for (int j = t; j < nper; j += 128) {
            float v = sv[j]; int cidx = si[j];
            if (v < bv || (v == bv && cidx < bc)) { bv = v; bj = j; bc = cidx; }
        }
        rv[t] = bv; ri[t] = bj;
        __syncthreads();
        for (int s = 64; s > 0; s >>= 1) {
            if (t < s) {
                float ov = rv[t + s]; int oj = ri[t + s];
                int oc = si[oj], mc = si[ri[t]];
                if (ov < rv[t] || (ov == rv[t] && oc < mc)) { rv[t] = ov; ri[t] = oj; }
            }
            __syncthreads();
        }
        if (t == 0) { int w = ri[0]; outIdx[q * 32 + k] = si[w]; sv[w] = 3.0e38f; }
        __syncthreads();
    }
}

// ---------------------------------------------------------------------------
// Gather: h[b*32+j] = concat(x_emb[b], cand_emb[idx[b][j]])  (1024 bf16/row)
// ---------------------------------------------------------------------------
__global__ void __launch_bounds__(256) k_gather(const __bf16* __restrict__ xe,
                                                const __bf16* __restrict__ ce,
                                                const int* __restrict__ idx,
                                                __bf16* __restrict__ h) {
    const int row = blockIdx.x;           // 0 .. B*32-1
    const int b = row >> 5, j = row & 31;
    const int id = idx[b * 32 + j];
    const int t = threadIdx.x;
    uint2* dst = (uint2*)(h + (size_t)row * 1024);
    if (t < 128) dst[t] = ((const uint2*)(xe + (size_t)b  * 512))[t];
    else         dst[t] = ((const uint2*)(ce + (size_t)id * 512))[t - 128];
}

// ---------------------------------------------------------------------------
// Final: out[row] = dot(h[row, 0:1024], wo) + bo   (one wave32 per row)
// ---------------------------------------------------------------------------
__global__ void __launch_bounds__(256) k_outdot(const __bf16* __restrict__ h,
                                                const __bf16* __restrict__ wo,
                                                const float* __restrict__ bo,
                                                float* __restrict__ out, int rows) {
    int wave = threadIdx.x >> 5, lane = threadIdx.x & 31;
    int row = blockIdx.x * 8 + wave;
    if (row >= rows) return;
    const __bf16* p = h + (size_t)row * 1024;
    float s = 0.f;
    for (int i = lane; i < 1024; i += 32) s += bf2f(p[i]) * bf2f(wo[i]);
    #pragma unroll
    for (int o = 16; o > 0; o >>= 1) s += __shfl_down(s, o, 32);
    if (lane == 0) out[row] = s + bo[0];
}

// ---------------------------------------------------------------------------
extern "C" void kernel_launch(void* const* d_in, const int* in_sizes, int n_in,
                              void* d_out, int out_size, void* d_ws, size_t ws_size,
                              hipStream_t stream) {
    (void)in_sizes; (void)n_in; (void)out_size; (void)ws_size;
    constexpr int B = 1024, N = 131072, F = 256, D = 512, D2 = 1024, TK = 32;

    const float* x   = (const float*)d_in[0];
    const float* cx  = (const float*)d_in[1];
    const float* W1  = (const float*)d_in[2];
    const float* b1  = (const float*)d_in[3];
    const float* g1  = (const float*)d_in[4];
    const float* be1 = (const float*)d_in[5];
    const float* m1  = (const float*)d_in[6];
    const float* v1  = (const float*)d_in[7];
    const float* W2  = (const float*)d_in[8];
    const float* b2  = (const float*)d_in[9];
    const float* W3  = (const float*)d_in[10];
    const float* b3  = (const float*)d_in[11];
    const float* g2  = (const float*)d_in[12];
    const float* be2 = (const float*)d_in[13];
    const float* m2  = (const float*)d_in[14];
    const float* v2  = (const float*)d_in[15];
    const float* Wm0 = (const float*)d_in[16];
    const float* bm0 = (const float*)d_in[17];
    const float* Wm1 = (const float*)d_in[18];
    const float* bm1 = (const float*)d_in[19];
    const float* Wo  = (const float*)d_in[20];
    const float* bo  = (const float*)d_in[21];
    float* out = (float*)d_out;

    // ---- workspace carve-out (256B aligned) ----
    char* ws = (char*)d_ws;
    size_t cur = 0;
    auto take = [&](size_t bytes) { char* p = ws + cur; cur += (bytes + 255) & ~(size_t)255; return p; };
    __bf16* cx_bf  = (__bf16*)take((size_t)N * F * 2);
    __bf16* x_bf   = (__bf16*)take((size_t)B * F * 2);
    __bf16* w1_bf  = (__bf16*)take((size_t)D * F * 2);
    __bf16* w2_bf  = (__bf16*)take((size_t)D * D * 2);
    __bf16* w3_bf  = (__bf16*)take((size_t)D * D * 2);
    __bf16* wm0_bf = (__bf16*)take((size_t)D2 * D2 * 2);
    __bf16* wm1_bf = (__bf16*)take((size_t)D2 * D2 * 2);
    __bf16* wo_bf  = (__bf16*)take((size_t)D2 * 2);
    __bf16* bufA   = (__bf16*)take((size_t)N * D * 2);   // candidate activations ping
    __bf16* bufB   = (__bf16*)take((size_t)N * D * 2);   // candidate activations pong
    __bf16* xeA    = (__bf16*)take((size_t)B * D * 2);
    __bf16* xeB    = (__bf16*)take((size_t)B * D * 2);
    float*  cn     = (float*) take((size_t)N * 4);
    float*  xn     = (float*) take((size_t)B * 4);
    float*  d2     = (float*) take((size_t)B * N * 4);
    float*  pv     = (float*) take((size_t)B * 32 * 32 * 4);
    int*    pi     = (int*)   take((size_t)B * 32 * 32 * 4);
    int*    tidx   = (int*)   take((size_t)B * 32 * 4);
    __bf16* h1     = (__bf16*)take((size_t)B * TK * D2 * 2);
    __bf16* h2     = (__bf16*)take((size_t)B * TK * D2 * 2);

    // ---- 1) fp32 -> bf16 conversions ----
    k_f2bf<<<4096, 256, 0, stream>>>(cx,  cx_bf,  (size_t)N * F);
    k_f2bf<<<512,  256, 0, stream>>>(x,   x_bf,   (size_t)B * F);
    k_f2bf<<<256,  256, 0, stream>>>(W1,  w1_bf,  (size_t)D * F);
    k_f2bf<<<512,  256, 0, stream>>>(W2,  w2_bf,  (size_t)D * D);
    k_f2bf<<<512,  256, 0, stream>>>(W3,  w3_bf,  (size_t)D * D);
    k_f2bf<<<1024, 256, 0, stream>>>(Wm0, wm0_bf, (size_t)D2 * D2);
    k_f2bf<<<1024, 256, 0, stream>>>(Wm1, wm1_bf, (size_t)D2 * D2);
    k_f2bf<<<4,    256, 0, stream>>>(Wo,  wo_bf,  (size_t)D2);

    // ---- 2) candidate embed: BN(W1) -> relu(W2) -> BN(W3) ----
    k_gemm<EP_BN, true><<<dim3(D / 128, N / 128), 256, 0, stream>>>(
        cx_bf, w1_bf, N, D, F, b1, g1, be1, m1, v1, nullptr, bufA);
    k_gemm<EP_BIAS_RELU, true><<<dim3(D / 128, N / 128), 256, 0, stream>>>(
        bufA, w2_bf, N, D, D, b2, nullptr, nullptr, nullptr, nullptr, nullptr, bufB);
    k_gemm<EP_BN, true><<<dim3(D / 128, N / 128), 256, 0, stream>>>(
        bufB, w3_bf, N, D, D, b3, g2, be2, m2, v2, nullptr, bufA);   // cand_emb = bufA

    // ---- 3) query embed (same weights, M = 1024) ----
    k_gemm<EP_BN, true><<<dim3(D / 128, B / 128), 256, 0, stream>>>(
        x_bf, w1_bf, B, D, F, b1, g1, be1, m1, v1, nullptr, xeA);
    k_gemm<EP_BIAS_RELU, true><<<dim3(D / 128, B / 128), 256, 0, stream>>>(
        xeA, w2_bf, B, D, D, b2, nullptr, nullptr, nullptr, nullptr, nullptr, xeB);
    k_gemm<EP_BN, true><<<dim3(D / 128, B / 128), 256, 0, stream>>>(
        xeB, w3_bf, B, D, D, b3, g2, be2, m2, v2, nullptr, xeA);     // x_emb = xeA

    // ---- 4) norms + distance GEMM: d2 = ||x||^2 - 2 x.c + ||c||^2 ----
    k_rownorm<<<N / 8, 256, 0, stream>>>(bufA, cn, N, D);
    k_rownorm<<<B / 8, 256, 0, stream>>>(xeA,  xn, B, D);
    k_gemm<EP_DIST, false><<<dim3(N / 128, B / 128), 256, 0, stream>>>(
        xeA, bufA, B, N, D, cn, nullptr, nullptr, nullptr, nullptr, xn, d2);

    // ---- 5) top-32 per query (chunked argmin-extract, then merge) ----
    k_topk_part<<<dim3(N / 4096, B), 128, 0, stream>>>(d2, N, pv, pi);
    k_topk_merge<<<B, 128, 0, stream>>>(pv, pi, (N / 4096) * 32, tidx);

    // ---- 6) gather context, head MLP, output ----
    k_gather<<<B * TK, 256, 0, stream>>>(xeA, bufA, tidx, h1);
    k_gemm<EP_BIAS_RELU, true><<<dim3(D2 / 128, (B * TK) / 128), 256, 0, stream>>>(
        h1, wm0_bf, B * TK, D2, D2, bm0, nullptr, nullptr, nullptr, nullptr, nullptr, h2);
    k_gemm<EP_BIAS_RELU, true><<<dim3(D2 / 128, (B * TK) / 128), 256, 0, stream>>>(
        h2, wm1_bf, B * TK, D2, D2, bm1, nullptr, nullptr, nullptr, nullptr, nullptr, h1);
    k_outdot<<<(B * TK) / 8, 256, 0, stream>>>(h1, wo_bf, bo, out, B * TK);
}
